// Head_29798483100238
// MI455X (gfx1250) — matrix operand order repaired and checked by reading, not compiled, not run on hardware
//
#include <hip/hip_runtime.h>
#include <hip/hip_bf16.h>

typedef __bf16 bf16;
typedef __attribute__((ext_vector_type(16))) __bf16 v16bf;
typedef __attribute__((ext_vector_type(8)))  __bf16 v8bf;
typedef __attribute__((ext_vector_type(8)))  float   v8f;

#define BB 32
#define TT 2048
#define CC 384
#define HH 64
#define BT (BB * TT)
#define ATTN_SCALE 0.05103103630798287f  // 384^-0.5

__device__ __forceinline__ bf16 f2bf(float f) { return (bf16)f; }

__device__ __forceinline__ v16bf cat(v8bf lo, v8bf hi) {
  return __builtin_shufflevector(lo, hi, 0, 1, 2, 3, 4, 5, 6, 7, 8, 9, 10, 11,
                                 12, 13, 14, 15);
}

__device__ __forceinline__ v8f wmma_bf16(v16bf a, v16bf b, v8f c) {
  return __builtin_amdgcn_wmma_f32_16x16x32_bf16(false, a, false, b, (short)0, c,
                                                 false, false);
}

// ---------------------------------------------------------------------------
// Kernel 1: QKV projections. grid = (BT/64, 3), block = 128 (4 waves).
// Wave computes a 16-row x 64-col output tile via 12 K-steps of bf16 WMMA.
// proj 0 -> k (row-major bf16), 1 -> q (row-major bf16, pre-scaled),
// proj 2 -> v stored transposed as vt[b][h][t] (bf16) for contiguous B-loads.
// ---------------------------------------------------------------------------
#define WT_STRIDE 392  // 384 + 8 pad; *2B = 784 = 49*16 -> rows stay 16B aligned

__global__ __launch_bounds__(128) void proj_kernel(
    const float* __restrict__ x, const float* __restrict__ Wk,
    const float* __restrict__ Wq, const float* __restrict__ Wv,
    bf16* __restrict__ kout, bf16* __restrict__ qout, bf16* __restrict__ vtout) {
  __shared__ bf16 wt[HH * WT_STRIDE];  // W^T in bf16: [h][c], padded

  const int tid  = threadIdx.x;
  const int proj = blockIdx.y;
  const float* W = (proj == 0) ? Wk : (proj == 1) ? Wq : Wv;
  const float scl = (proj == 1) ? ATTN_SCALE : 1.0f;

  for (int idx = tid; idx < CC * HH; idx += 128) {
    int c = idx >> 6, h = idx & 63;
    wt[h * WT_STRIDE + c] = f2bf(W[idx] * scl);
  }
  __syncthreads();

  const int lane = tid & 31, wid = tid >> 5;
  const int half = lane >> 4, l15 = lane & 15;
  const int r0 = (blockIdx.x * 4 + wid) * 16;

  v8f acc[4];
#pragma unroll
  for (int n = 0; n < 4; ++n) acc[n] = (v8f){0.f,0.f,0.f,0.f,0.f,0.f,0.f,0.f};

  // A-fragment base: row = r0 + l15, K base per half = half*8 (+16 for hi seg)
  const float* xr = x + (size_t)(r0 + l15) * CC + half * 8;

#pragma unroll
  for (int step = 0; step < CC / 32; ++step) {
    const float* xs = xr + step * 32;
    v8f fa = *(const v8f*)(xs);        // K = half*8 + {0..7}
    v8f fb = *(const v8f*)(xs + 16);   // K = 16 + half*8 + {0..7}
    v16bf xa = cat(__builtin_convertvector(fa, v8bf),
                   __builtin_convertvector(fb, v8bf));

#pragma unroll
    for (int n = 0; n < 4; ++n) {
      // B-fragment: N = n*16 + l15 (head col), K = step*32 + half*8 + {0..7}, +16
      const v8bf* wp =
          (const v8bf*)&wt[(n * 16 + l15) * WT_STRIDE + step * 32 + half * 8];
      v16bf wb = cat(wp[0], wp[2]);
      acc[n] = wmma_bf16(xa, wb, acc[n]);
    }
  }

  if (proj < 2) {
    bf16* ob = ((proj == 0) ? kout : qout) + (size_t)r0 * HH;
#pragma unroll
    for (int n = 0; n < 4; ++n)
#pragma unroll
      for (int i = 0; i < 8; ++i)
        ob[(size_t)(half * 8 + i) * HH + n * 16 + l15] = f2bf(acc[n][i]);
  } else {
    const int b = r0 >> 11;           // r0 / TT
    const int t0 = r0 & (TT - 1);
#pragma unroll
    for (int n = 0; n < 4; ++n)
#pragma unroll
      for (int i = 0; i < 8; ++i)
        vtout[((size_t)b * HH + n * 16 + l15) * TT + t0 + half * 8 + i] =
            f2bf(acc[n][i]);
  }
}

// ---------------------------------------------------------------------------
// Kernel 2: fused causal flash attention. grid = BT/64, block = 128 (4 waves).
// Each wave owns 16 query rows; iterates 32-key tiles up to the diagonal.
// ---------------------------------------------------------------------------
#define P_STRIDE 40  // 32 + 8 pad; *2B = 80 -> 16B-aligned A-fragment reloads

__global__ __launch_bounds__(128) void flash_kernel(
    const bf16* __restrict__ kbuf, const bf16* __restrict__ qbuf,
    const bf16* __restrict__ vtbuf, float* __restrict__ out) {
  __shared__ bf16 pbuf[4 * 16 * P_STRIDE];

  const int tid = threadIdx.x;
  const int lane = tid & 31, wid = tid >> 5;
  const int half = lane >> 4, l15 = lane & 15;

  const int rowTile = blockIdx.x * 4 + wid;  // 16-row tile index over B*T
  const int b  = rowTile >> 7;               // 128 tiles per batch
  const int m0 = (rowTile & 127) << 4;       // first query row in batch

  const bf16* qb = qbuf + ((size_t)b * TT + m0) * HH;
  const bf16* kb = kbuf + (size_t)b * TT * HH;
  const bf16* vb = vtbuf + (size_t)b * HH * TT;

  // Q A-fragments (head chunks of 32), kept in registers for the whole loop.
  v16bf qa[2];
#pragma unroll
  for (int c = 0; c < 2; ++c) {
    const v8bf* p0 = (const v8bf*)(qb + (size_t)l15 * HH + c * 32 + half * 8);
    const v8bf* p1 = (const v8bf*)(qb + (size_t)l15 * HH + c * 32 + half * 8 + 16);
    qa[c] = cat(*p0, *p1);
  }

  v8f o[4];
#pragma unroll
  for (int n = 0; n < 4; ++n) o[n] = (v8f){0.f,0.f,0.f,0.f,0.f,0.f,0.f,0.f};
  float m8[8], l8[8];
#pragma unroll
  for (int i = 0; i < 8; ++i) { m8[i] = -1e30f; l8[i] = 0.f; }

  bf16* pl = pbuf + wid * 16 * P_STRIDE;
  const int rbase = m0 + half * 8;  // row of element i is rbase + i

  for (int kt = 0; kt < m0 + 16; kt += 32) {
    // K B-fragments: N = key = kt + t*16 + l15, K-dim = head = c*32 + half*8 + {0..7}, +16
    v16bf kf[2][2];
#pragma unroll
    for (int t = 0; t < 2; ++t)
#pragma unroll
      for (int c = 0; c < 2; ++c) {
        const v8bf* kp =
            (const v8bf*)(kb + (size_t)(kt + t * 16 + l15) * HH + c * 32 + half * 8);
        kf[t][c] = cat(kp[0], kp[2]);
      }

    v8f s0 = (v8f){0.f,0.f,0.f,0.f,0.f,0.f,0.f,0.f};
    v8f s1 = (v8f){0.f,0.f,0.f,0.f,0.f,0.f,0.f,0.f};
    s0 = wmma_bf16(qa[0], kf[0][0], s0);
    s0 = wmma_bf16(qa[1], kf[0][1], s0);
    s1 = wmma_bf16(qa[0], kf[1][0], s1);
    s1 = wmma_bf16(qa[1], kf[1][1], s1);

    // Causal mask (scores already scaled via pre-scaled Q).
    float p0[8], p1[8];
    const int k0 = kt + l15, k1 = kt + 16 + l15;
#pragma unroll
    for (int i = 0; i < 8; ++i) {
      int r = rbase + i;
      p0[i] = (k0 <= r) ? s0[i] : -1e30f;
      p1[i] = (k1 <= r) ? s1[i] : -1e30f;
    }

    // Row-wise max across the 16 lanes of each half-wave.
    float mx[8];
#pragma unroll
    for (int i = 0; i < 8; ++i) mx[i] = fmaxf(p0[i], p1[i]);
#pragma unroll
    for (int off = 8; off >= 1; off >>= 1)
#pragma unroll
      for (int i = 0; i < 8; ++i)
        mx[i] = fmaxf(mx[i], __shfl_xor(mx[i], off, 32));

    float alpha[8];
#pragma unroll
    for (int i = 0; i < 8; ++i) {
      float mn = fmaxf(m8[i], mx[i]);
      alpha[i] = __expf(m8[i] - mn);
      m8[i] = mn;
    }
    float rs[8];
#pragma unroll
    for (int i = 0; i < 8; ++i) {
      p0[i] = __expf(p0[i] - m8[i]);
      p1[i] = __expf(p1[i] - m8[i]);
      rs[i] = p0[i] + p1[i];
    }
#pragma unroll
    for (int off = 8; off >= 1; off >>= 1)
#pragma unroll
      for (int i = 0; i < 8; ++i) rs[i] += __shfl_xor(rs[i], off, 32);
#pragma unroll
    for (int i = 0; i < 8; ++i) l8[i] = l8[i] * alpha[i] + rs[i];
#pragma unroll
    for (int n = 0; n < 4; ++n)
#pragma unroll
      for (int i = 0; i < 8; ++i) o[n][i] *= alpha[i];

    // P (C-layout) -> LDS -> A-layout bf16 fragment. Wave-private buffer.
#pragma unroll
    for (int i = 0; i < 8; ++i) {
      pl[(half * 8 + i) * P_STRIDE + l15]      = f2bf(p0[i]);
      pl[(half * 8 + i) * P_STRIDE + 16 + l15] = f2bf(p1[i]);
    }
    asm volatile("s_wait_dscnt 0" ::: "memory");

    v16bf pa;
    {
      const v8bf* a0 = (const v8bf*)(pl + l15 * P_STRIDE + half * 8);
      const v8bf* a1 = (const v8bf*)(pl + l15 * P_STRIDE + half * 8 + 16);
      pa = cat(*a0, *a1);
    }

    // PV: B-fragment from head-major vt -> two 16B chunks per lane, 16 keys apart.
#pragma unroll
    for (int n = 0; n < 4; ++n) {
      const v8bf* vp =
          (const v8bf*)(vb + (size_t)(n * 16 + l15) * TT + kt + half * 8);
      v16bf vf = cat(vp[0], vp[2]);
      o[n] = wmma_bf16(pa, vf, o[n]);
    }
  }

  float* ob = out + ((size_t)b * TT + m0) * HH;
#pragma unroll
  for (int i = 0; i < 8; ++i) {
    float inv = 1.0f / l8[i];
#pragma unroll
    for (int n = 0; n < 4; ++n)
      ob[(size_t)(half * 8 + i) * HH + n * 16 + l15] = o[n][i] * inv;
  }
}

extern "C" void kernel_launch(void* const* d_in, const int* in_sizes, int n_in,
                              void* d_out, int out_size, void* d_ws, size_t ws_size,
                              hipStream_t stream) {
  (void)in_sizes; (void)n_in; (void)out_size; (void)ws_size;
  const float* x  = (const float*)d_in[0];
  const float* Wk = (const float*)d_in[1];
  const float* Wq = (const float*)d_in[2];
  const float* Wv = (const float*)d_in[3];

  // Workspace: k, q (row-major bf16) and v transposed (head-major bf16).
  bf16* kbuf  = (bf16*)d_ws;
  bf16* qbuf  = kbuf + (size_t)BT * HH;
  bf16* vtbuf = qbuf + (size_t)BT * HH;

  proj_kernel<<<dim3(BT / 64, 3), 128, 0, stream>>>(x, Wk, Wq, Wv, kbuf, qbuf,
                                                    vtbuf);
  flash_kernel<<<dim3(BT / 64), 128, 0, stream>>>(kbuf, qbuf, vtbuf,
                                                  (float*)d_out);
}
